// VRNN_11278584119715
// MI455X (gfx1250) — matrix operand order repaired and checked by reading, not compiled, not run on hardware
//
#include <hip/hip_runtime.h>
#include <hip/hip_bf16.h>
#include <math.h>

// ---------------------------------------------------------------------------
// Types for CDNA5 WMMA (wave32): V_WMMA_F32_16X16X32_F16
// ---------------------------------------------------------------------------
typedef __attribute__((ext_vector_type(16))) _Float16 v16h;
typedef __attribute__((ext_vector_type(8)))  _Float16 v8h;
typedef __attribute__((ext_vector_type(8)))  float    v8f;

static __device__ __forceinline__ v8f wmma_f16(v16h a, v16h b, v8f c) {
  // 8 args: (neg_a, A, neg_b, B, c_mod, C, reuse_a, reuse_b)
  return __builtin_amdgcn_wmma_f32_16x16x32_f16(false, a, false, b, (short)0, c,
                                                false, false);
}

// A fragment (16x32, f16, row-major source, leading dim ld):
// lanes 0-15: row=lane, K = k0..k0+15 contiguous
// lanes 16-31: row=lane-16, K = k0+16..k0+31 contiguous
static __device__ __forceinline__ v16h load_a_frag(const _Float16* A, int ld,
                                                   int row0, int k0, int lane) {
  int r = row0 + (lane & 15);
  int hsel = (lane >> 4) ? 16 : 0;
  const _Float16* p = A + (size_t)r * ld + k0 + hsel;
  v8h lo = *(const v8h*)(p);
  v8h hi = *(const v8h*)(p + 8);
  v16h a;
#pragma unroll
  for (int i = 0; i < 8; ++i) { a[i] = lo[i]; a[i + 8] = hi[i]; }
  return a;
}

// B fragment (32x16): B[k][n] = W[n][k] with W row-major [N][K].
// lane n (0..15): column n, K=k0..k0+15 contiguous; lane n+16: K=k0+16..k0+31.
static __device__ __forceinline__ v16h load_b_frag(const _Float16* W, int ld,
                                                   int n0, int k0, int lane) {
  int n = n0 + (lane & 15);
  int koff = k0 + ((lane >> 4) ? 16 : 0);
  return *(const v16h*)(W + (size_t)n * ld + koff);
}

static __device__ __forceinline__ float sigmoidf_(float x) {
  return 1.0f / (1.0f + expf(-x));
}

// ---------------------------------------------------------------------------
// Constants
// ---------------------------------------------------------------------------
#define LSTEPS 64
#define BATCH  32
#define HID    256
#define LAT    256
#define VOC    32000
#define ROWS   (LSTEPS * BATCH)   // 2048

// ---------------------------------------------------------------------------
// f32 -> f16 convert
// ---------------------------------------------------------------------------
__global__ void k_f32_to_f16(const float* __restrict__ src,
                             _Float16* __restrict__ dst, int n) {
  int i = blockIdx.x * blockDim.x + threadIdx.x;
  if (i < n) dst[i] = (_Float16)src[i];
}

// Embedding gather -> f16 x[L*B, H]
__global__ void k_gather_emb(const int* __restrict__ nodes,
                             const float* __restrict__ emb,
                             _Float16* __restrict__ x16) {
  int i = blockIdx.x * blockDim.x + threadIdx.x;
  if (i >= ROWS * HID) return;
  int row = i >> 8, k = i & 255;
  x16[i] = (_Float16)emb[(size_t)nodes[row] * HID + k];
}

// ---------------------------------------------------------------------------
// gi GEMM: gi[cell][L*B][3H] = x @ Wih_cell.T + bih_cell  (4 cells fused, WMMA)
// N = 4*768 = 3072 -> 192 tiles, M = 2048 -> 128 tiles
// ---------------------------------------------------------------------------
__global__ __launch_bounds__(256) void k_gi_gemm(
    const _Float16* __restrict__ x16, const _Float16* __restrict__ wih16,
    const float* __restrict__ bih_ec, const float* __restrict__ bih_es,
    const float* __restrict__ bih_dc, const float* __restrict__ bih_ds,
    float* __restrict__ gi) {
  int lane = threadIdx.x & 31;
  int wave = threadIdx.x >> 5;
  int ntile = blockIdx.x * 8 + wave;   // 0..191
  int mtile = blockIdx.y;              // 0..127
  int cell = ntile / 48;               // 768/16 = 48 tiles per cell
  int jt = ntile % 48;
  const _Float16* W = wih16 + (size_t)cell * 768 * HID;
  const float* bih = (cell == 0) ? bih_ec : (cell == 1) ? bih_es
                   : (cell == 2) ? bih_dc : bih_ds;
  v8f acc = {};
#pragma unroll
  for (int k0 = 0; k0 < HID; k0 += 32) {
    v16h a = load_a_frag(x16, HID, mtile * 16, k0, lane);
    v16h b = load_b_frag(W, HID, jt * 16, k0, lane);
    acc = wmma_f16(a, b, acc);
  }
  int col = jt * 16 + (lane & 15);
  float bv = bih[col];
  float* out = gi + (size_t)cell * ROWS * 768 + (size_t)mtile * 16 * 768;
  int rbase = (lane >> 4) * 8;
#pragma unroll
  for (int v = 0; v < 8; ++v) out[(size_t)(rbase + v) * 768 + col] = acc[v] + bv;
}

// ---------------------------------------------------------------------------
// Persistent tree-GRU scan: one workgroup (8 waves), 64 sequential steps.
// Each wave owns 4 (mtile,jtile) output tiles; per tile it accumulates the
// r/z/n gate tiles for BOTH cells (6 WMMA chains, shared A fragment), then
// finishes the GRU elementwise math locally and applies the edge select.
// h kept in LDS in f32 (for z*h) and f16 (WMMA A operand).
// ---------------------------------------------------------------------------
__global__ __launch_bounds__(256) void k_tree_scan(
    const float* __restrict__ h0, const _Float16* __restrict__ whh_c,
    const _Float16* __restrict__ whh_s, const float* __restrict__ bhh_c,
    const float* __restrict__ bhh_s, const float* __restrict__ gi_c,
    const float* __restrict__ gi_s, const unsigned char* __restrict__ edges,
    _Float16* __restrict__ outs16 /* nullable */,
    float* __restrict__ h_final /* nullable */) {
  __shared__ __align__(32) float    h32[BATCH * HID];   // 32 KB
  __shared__ __align__(32) _Float16 h16[BATCH * HID];   // 16 KB
  int tid = threadIdx.x, lane = tid & 31, wave = tid >> 5;

  for (int i = tid; i < BATCH * HID; i += 256) {
    float v = h0[i];
    h32[i] = v;
    h16[i] = (_Float16)v;
  }
  __syncthreads();

  for (int t = 0; t < LSTEPS; ++t) {
    const float* gc = gi_c + (size_t)t * BATCH * 768;
    const float* gs = gi_s + (size_t)t * BATCH * 768;
    float hnew[4][8];
#pragma unroll
    for (int p = 0; p < 4; ++p) {
      int pair = wave * 4 + p;   // 0..31 = 2 mtiles x 16 jtiles
      int m = pair & 1;
      int jt = pair >> 1;
      v8f aR = {}, aZ = {}, aN = {}, sR = {}, sZ = {}, sN = {};
#pragma unroll
      for (int k0 = 0; k0 < HID; k0 += 32) {
        v16h a = load_a_frag(h16, HID, m * 16, k0, lane);
        v16h bcr = load_b_frag(whh_c, HID, jt * 16, k0, lane);
        v16h bcz = load_b_frag(whh_c, HID, jt * 16 + 256, k0, lane);
        v16h bcn = load_b_frag(whh_c, HID, jt * 16 + 512, k0, lane);
        v16h bsr = load_b_frag(whh_s, HID, jt * 16, k0, lane);
        v16h bsz = load_b_frag(whh_s, HID, jt * 16 + 256, k0, lane);
        v16h bsn = load_b_frag(whh_s, HID, jt * 16 + 512, k0, lane);
        aR = wmma_f16(a, bcr, aR);
        aZ = wmma_f16(a, bcz, aZ);
        aN = wmma_f16(a, bcn, aN);
        sR = wmma_f16(a, bsr, sR);
        sZ = wmma_f16(a, bsz, sZ);
        sN = wmma_f16(a, bsn, sN);
      }
      int col = jt * 16 + (lane & 15);
      int rbase = (lane >> 4) * 8;
      float bcr_ = bhh_c[col], bcz_ = bhh_c[col + 256], bcn_ = bhh_c[col + 512];
      float bsr_ = bhh_s[col], bsz_ = bhh_s[col + 256], bsn_ = bhh_s[col + 512];
#pragma unroll
      for (int v = 0; v < 8; ++v) {
        int b = m * 16 + rbase + v;
        float hprev = h32[b * HID + col];
        const float* gcr = gc + (size_t)b * 768;
        const float* gsr = gs + (size_t)b * 768;
        float rc = sigmoidf_(gcr[col] + aR[v] + bcr_);
        float zc = sigmoidf_(gcr[col + 256] + aZ[v] + bcz_);
        float nc = tanhf(gcr[col + 512] + rc * (aN[v] + bcn_));
        float hc = (1.0f - zc) * nc + zc * hprev;
        float rs = sigmoidf_(gsr[col] + sR[v] + bsr_);
        float zs = sigmoidf_(gsr[col + 256] + sZ[v] + bsz_);
        float ns = tanhf(gsr[col + 512] + rs * (sN[v] + bsn_));
        float hs = (1.0f - zs) * ns + zs * hprev;
        hnew[p][v] = edges[t * BATCH + b] ? hc : hs;
      }
    }
    __syncthreads();  // all reads of h done before writes
#pragma unroll
    for (int p = 0; p < 4; ++p) {
      int pair = wave * 4 + p;
      int m = pair & 1;
      int jt = pair >> 1;
      int col = jt * 16 + (lane & 15);
      int rbase = (lane >> 4) * 8;
#pragma unroll
      for (int v = 0; v < 8; ++v) {
        int b = m * 16 + rbase + v;
        float x = hnew[p][v];
        h32[b * HID + col] = x;
        _Float16 xh = (_Float16)x;
        h16[b * HID + col] = xh;
        if (outs16) outs16[(size_t)(t * BATCH + b) * HID + col] = xh;
      }
    }
    __syncthreads();
  }
  if (h_final)
    for (int i = tid; i < BATCH * HID; i += 256) h_final[i] = h32[i];
}

// ---------------------------------------------------------------------------
// VAE: mean/logv/z (stage1), h0 = z @ W_l2h.T + b (stage2). Tiny; plain VALU.
// ---------------------------------------------------------------------------
__global__ void k_vae1(const float* __restrict__ h_enc,
                       const float* __restrict__ Wm, const float* __restrict__ bm,
                       const float* __restrict__ Wl, const float* __restrict__ bl,
                       const float* __restrict__ eps, float* __restrict__ o_mean,
                       float* __restrict__ o_logv, float* __restrict__ o_z,
                       float* __restrict__ z_ws) {
  int i = blockIdx.x * blockDim.x + threadIdx.x;
  if (i >= BATCH * LAT) return;
  int b = i >> 8, zc = i & 255;
  const float* h = h_enc + (size_t)b * HID;
  const float* wm = Wm + (size_t)zc * HID;
  const float* wl = Wl + (size_t)zc * HID;
  float am = 0.0f, al = 0.0f;
  for (int k = 0; k < HID; ++k) { float hv = h[k]; am += hv * wm[k]; al += hv * wl[k]; }
  am += bm[zc]; al += bl[zc];
  float zv = eps[i] * expf(0.5f * al) + am;
  o_mean[i] = am; o_logv[i] = al; o_z[i] = zv; z_ws[i] = zv;
}

__global__ void k_vae2(const float* __restrict__ z, const float* __restrict__ W,
                       const float* __restrict__ bias, float* __restrict__ h0) {
  int i = blockIdx.x * blockDim.x + threadIdx.x;
  if (i >= BATCH * HID) return;
  int b = i >> 8, j = i & 255;
  const float* zr = z + (size_t)b * LAT;
  const float* wr = W + (size_t)j * LAT;
  float a = bias[j];
  for (int k = 0; k < LAT; ++k) a += zr[k] * wr[k];
  h0[i] = a;
}

// ---------------------------------------------------------------------------
// Vocab GEMM: logits[2048, 32000] = outs16 @ Wvoc16.T + b_voc   (WMMA)
// 2x2 register blocking per wave: 2 A-frags + 2 B-frags -> 4 WMMAs per k-step
// (doubles L2 arithmetic intensity vs 1x1). Grid exact: 64 M-pairs x 1000
// N-pairs, 8 waves/block -> dim3(125, 64).
// ---------------------------------------------------------------------------
__global__ __launch_bounds__(256) void k_vocab_gemm(
    const _Float16* __restrict__ outs16, const _Float16* __restrict__ wvoc16,
    const float* __restrict__ bvoc, float* __restrict__ logits) {
  int lane = threadIdx.x & 31;
  int wave = threadIdx.x >> 5;
  int n0 = (blockIdx.x * 8 + wave) * 32;   // N-pair base, 0..31968
  int m0 = blockIdx.y * 32;                // M-pair base, 0..2016
  v8f acc00 = {}, acc01 = {}, acc10 = {}, acc11 = {};
#pragma unroll
  for (int k0 = 0; k0 < HID; k0 += 32) {
    v16h a0 = load_a_frag(outs16, HID, m0, k0, lane);
    v16h a1 = load_a_frag(outs16, HID, m0 + 16, k0, lane);
    v16h b0 = load_b_frag(wvoc16, HID, n0, k0, lane);
    v16h b1 = load_b_frag(wvoc16, HID, n0 + 16, k0, lane);
    acc00 = wmma_f16(a0, b0, acc00);
    acc01 = wmma_f16(a0, b1, acc01);
    acc10 = wmma_f16(a1, b0, acc10);
    acc11 = wmma_f16(a1, b1, acc11);
  }
  int cl = lane & 15;
  int rbase = (lane >> 4) * 8;
  float bv0 = bvoc[n0 + cl];
  float bv1 = bvoc[n0 + 16 + cl];
#pragma unroll
  for (int v = 0; v < 8; ++v) {
    size_t r0 = (size_t)(m0 + rbase + v) * VOC;
    size_t r1 = (size_t)(m0 + 16 + rbase + v) * VOC;
    logits[r0 + n0 + cl]      = acc00[v] + bv0;
    logits[r0 + n0 + 16 + cl] = acc01[v] + bv1;
    logits[r1 + n0 + cl]      = acc10[v] + bv0;
    logits[r1 + n0 + 16 + cl] = acc11[v] + bv1;
  }
}

// ---------------------------------------------------------------------------
// log-softmax over V per row (float4 vectorized): stats then apply in place.
// ---------------------------------------------------------------------------
__global__ void k_smax_stats(const float* __restrict__ logits,
                             float* __restrict__ stats) {
  __shared__ float red[256];
  int row = blockIdx.x;
  const float4* x = (const float4*)(logits + (size_t)row * VOC);  // 8000 f4
  float m = -3.4e38f;
  for (int i = threadIdx.x; i < VOC / 4; i += 256) {
    float4 v = x[i];
    m = fmaxf(m, fmaxf(fmaxf(v.x, v.y), fmaxf(v.z, v.w)));
  }
  red[threadIdx.x] = m;
  __syncthreads();
  for (int s = 128; s > 0; s >>= 1) {
    if (threadIdx.x < s) red[threadIdx.x] = fmaxf(red[threadIdx.x], red[threadIdx.x + s]);
    __syncthreads();
  }
  m = red[0];
  __syncthreads();
  float sum = 0.0f;
  for (int i = threadIdx.x; i < VOC / 4; i += 256) {
    float4 v = x[i];
    sum += expf(v.x - m) + expf(v.y - m) + expf(v.z - m) + expf(v.w - m);
  }
  red[threadIdx.x] = sum;
  __syncthreads();
  for (int s = 128; s > 0; s >>= 1) {
    if (threadIdx.x < s) red[threadIdx.x] += red[threadIdx.x + s];
    __syncthreads();
  }
  if (threadIdx.x == 0) stats[row] = m + logf(red[0]);
}

__global__ void k_smax_apply(float* __restrict__ logits,
                             const float* __restrict__ stats) {
  int row = blockIdx.x;
  float s = stats[row];
  float4* x = (float4*)(logits + (size_t)row * VOC);
  for (int i = threadIdx.x; i < VOC / 4; i += 256) {
    float4 v = x[i];
    v.x -= s; v.y -= s; v.z -= s; v.w -= s;
    x[i] = v;
  }
}

// ---------------------------------------------------------------------------
// Host launcher
// ---------------------------------------------------------------------------
extern "C" void kernel_launch(void* const* d_in, const int* in_sizes, int n_in,
                              void* d_out, int out_size, void* d_ws, size_t ws_size,
                              hipStream_t stream) {
  (void)in_sizes; (void)n_in; (void)out_size; (void)ws_size;
  // Inputs (setup_inputs order)
  const int*   nodes    = (const int*)d_in[0];
  const unsigned char* edges = (const unsigned char*)d_in[1];  // jnp bool -> 1 byte
  const float* enc_init = (const float*)d_in[2];
  const float* eps      = (const float*)d_in[3];
  const float* emb      = (const float*)d_in[4];
  const float* Wih[4] = {(const float*)d_in[5],  (const float*)d_in[9],
                         (const float*)d_in[13], (const float*)d_in[17]};
  const float* Whh[4] = {(const float*)d_in[6],  (const float*)d_in[10],
                         (const float*)d_in[14], (const float*)d_in[18]};
  const float* bih[4] = {(const float*)d_in[7],  (const float*)d_in[11],
                         (const float*)d_in[15], (const float*)d_in[19]};
  const float* bhh[4] = {(const float*)d_in[8],  (const float*)d_in[12],
                         (const float*)d_in[16], (const float*)d_in[20]};
  const float* W_mean = (const float*)d_in[21]; const float* b_mean = (const float*)d_in[22];
  const float* W_logv = (const float*)d_in[23]; const float* b_logv = (const float*)d_in[24];
  const float* W_l2h  = (const float*)d_in[25]; const float* b_l2h  = (const float*)d_in[26];
  const float* W_voc  = (const float*)d_in[27]; const float* b_voc  = (const float*)d_in[28];

  // Workspace layout (256B aligned slabs)
  char* ws = (char*)d_ws;
  size_t off = 0;
  auto alloc = [&](size_t bytes) -> char* {
    char* p = ws + off;
    off = (off + bytes + 255) & ~(size_t)255;
    return p;
  };
  _Float16* x16     = (_Float16*)alloc((size_t)ROWS * HID * 2);
  _Float16* wih16   = (_Float16*)alloc((size_t)4 * 768 * HID * 2);
  _Float16* whh16   = (_Float16*)alloc((size_t)4 * 768 * HID * 2);
  _Float16* wvoc16  = (_Float16*)alloc((size_t)VOC * HID * 2);
  float*    gi      = (float*)alloc((size_t)4 * ROWS * 768 * 4);
  float*    h_enc   = (float*)alloc((size_t)BATCH * HID * 4);
  float*    z_ws    = (float*)alloc((size_t)BATCH * LAT * 4);
  float*    h0_dec  = (float*)alloc((size_t)BATCH * HID * 4);
  _Float16* outs16  = (_Float16*)alloc((size_t)ROWS * HID * 2);
  float*    stats   = (float*)alloc((size_t)ROWS * 4);

  // Output layout: logp [2048*32000] then mean, logv, z (each 8192 f32)
  float* logp   = (float*)d_out;
  float* o_mean = logp + (size_t)ROWS * VOC;
  float* o_logv = o_mean + BATCH * LAT;
  float* o_z    = o_logv + BATCH * LAT;

  const int CT = 256;
  // 1) weight converts + gather
  for (int c = 0; c < 4; ++c) {
    int n = 768 * HID;
    k_f32_to_f16<<<(n + CT - 1) / CT, CT, 0, stream>>>(Wih[c], wih16 + (size_t)c * n, n);
    k_f32_to_f16<<<(n + CT - 1) / CT, CT, 0, stream>>>(Whh[c], whh16 + (size_t)c * n, n);
  }
  {
    int n = VOC * HID;
    k_f32_to_f16<<<(n + CT - 1) / CT, CT, 0, stream>>>(W_voc, wvoc16, n);
  }
  k_gather_emb<<<(ROWS * HID + CT - 1) / CT, CT, 0, stream>>>(nodes, emb, x16);

  // 2) fused input-projection GEMM for all 4 cells (WMMA)
  k_gi_gemm<<<dim3(24, 128), 256, 0, stream>>>(x16, wih16, bih[0], bih[1],
                                               bih[2], bih[3], gi);

  // 3) encoder scan (persistent single workgroup, WMMA)
  k_tree_scan<<<1, 256, 0, stream>>>(
      enc_init, whh16 + (size_t)0 * 768 * HID, whh16 + (size_t)1 * 768 * HID,
      bhh[0], bhh[1], gi + (size_t)0 * ROWS * 768, gi + (size_t)1 * ROWS * 768,
      edges, ((_Float16*)nullptr), h_enc);

  // 4) VAE
  k_vae1<<<(BATCH * LAT + CT - 1) / CT, CT, 0, stream>>>(
      h_enc, W_mean, b_mean, W_logv, b_logv, eps, o_mean, o_logv, o_z, z_ws);
  k_vae2<<<(BATCH * HID + CT - 1) / CT, CT, 0, stream>>>(z_ws, W_l2h, b_l2h, h0_dec);

  // 5) decoder scan (writes f16 outs for vocab GEMM)
  k_tree_scan<<<1, 256, 0, stream>>>(
      h0_dec, whh16 + (size_t)2 * 768 * HID, whh16 + (size_t)3 * 768 * HID,
      bhh[2], bhh[3], gi + (size_t)2 * ROWS * 768, gi + (size_t)3 * ROWS * 768,
      edges, outs16, (float*)nullptr);

  // 6) vocab GEMM (WMMA, 2x2 register blocked): 2048 x 32000 x 256, bias fused
  k_vocab_gemm<<<dim3(125, 64), 256, 0, stream>>>(outs16, wvoc16, b_voc, logp);

  // 7) log-softmax in place (float4 vectorized)
  k_smax_stats<<<ROWS, 256, 0, stream>>>(logp, stats);
  k_smax_apply<<<ROWS, 256, 0, stream>>>(logp, stats);
}
